// TL_20255065767964
// MI455X (gfx1250) — matrix operand order repaired and checked by reading, not compiled, not run on hardware
//
#include <hip/hip_runtime.h>
#include <hip/hip_bf16.h>

// ---------------------------------------------------------------------------
// Ternary-quantized linear:  out[b,s,o] = sum_i x[b,s,i] * tq(W)[o,i] + b[o]
// GEMM: M = 32768, N = 1024, K = 1024.  bf16 WMMA, 2-term x split,
// async global->LDS double-buffered weight staging (gfx1250 ASYNCcnt path).
// ---------------------------------------------------------------------------

typedef __attribute__((ext_vector_type(16))) __bf16 v16bf;
typedef __attribute__((ext_vector_type(8)))  float  v8f;

#define KTOT 1024
#define NTOT 1024

// ------------------------- alpha = mean(|W|) kernels ------------------------

__global__ __launch_bounds__(256) void abssum_partial(const float* __restrict__ W,
                                                      float* __restrict__ partial,
                                                      int n) {
    __shared__ float sm[256];
    float s = 0.f;
    for (int i = blockIdx.x * 256 + threadIdx.x; i < n; i += gridDim.x * 256)
        s += fabsf(W[i]);
    sm[threadIdx.x] = s;
    __syncthreads();
    for (int off = 128; off > 0; off >>= 1) {
        if ((int)threadIdx.x < off) sm[threadIdx.x] += sm[threadIdx.x + off];
        __syncthreads();
    }
    if (threadIdx.x == 0) partial[blockIdx.x] = sm[0];
}

__global__ __launch_bounds__(256) void finalize_thresh(const float* __restrict__ partial,
                                                       float* __restrict__ thresh,
                                                       float inv_n) {
    __shared__ float sm[256];
    sm[threadIdx.x] = partial[threadIdx.x];   // exactly 256 partials
    __syncthreads();
    for (int off = 128; off > 0; off >>= 1) {
        if ((int)threadIdx.x < off) sm[threadIdx.x] += sm[threadIdx.x + off];
        __syncthreads();
    }
    if (threadIdx.x == 0) thresh[0] = sm[0] * inv_n * 0.5f;   // 0.5 * mean|W|
}

// wt = sign(w) * (|w| > 0.5*alpha)  stored as bf16 bits {0xBF80, 0, 0x3F80}
__global__ __launch_bounds__(256) void quantize_w(const float* __restrict__ W,
                                                  const float* __restrict__ thresh,
                                                  unsigned short* __restrict__ Wt,
                                                  int n) {
    const float t = thresh[0];
    int i = blockIdx.x * 256 + threadIdx.x;
    if (i < n) {
        float w = W[i];
        unsigned short q = 0;
        if (fabsf(w) > t) q = (w > 0.f) ? (unsigned short)0x3F80 : (unsigned short)0xBF80;
        Wt[i] = q;
    }
}

// ------------------------------- GEMM kernel --------------------------------
// Block: 256 threads = 8 waves. Tile: 128(M) x 64(N) per block.
// Weights double-buffered in LDS via GLOBAL_LOAD_ASYNC_TO_LDS_B128.

__global__ __launch_bounds__(256) void gemm_ternary(const float* __restrict__ X,
                                                    const unsigned short* __restrict__ Wt,
                                                    const float* __restrict__ bias,
                                                    float* __restrict__ Out) {
    // B chunk: 64 cols x 32 K of bf16 = 4KB, двух-buffered = 8KB LDS.
    __shared__ __align__(16) unsigned short Bsm[2][64 * 32];

    const int lane = threadIdx.x & 31;
    const int wave = threadIdx.x >> 5;
    const int half = lane >> 4;      // 0: lanes 0-15, 1: lanes 16-31
    const int l15  = lane & 15;

    const int m0 = blockIdx.y * 128 + wave * 16;
    const int n0 = blockIdx.x * 64;

    // ---- async-staging mapping: thread t copies 16B: col = t/4, part = t%4 ----
    const int sc = threadIdx.x >> 2;                         // 0..63 (col in tile)
    const int sp = threadIdx.x & 3;                          // 0..3  (16B part of 64B row)
    const unsigned lds_in = (unsigned)(sc * 64 + sp * 16);   // bytes within chunk
    const unsigned lbase0 = (unsigned)(uintptr_t)&Bsm[0][0]; // raw LDS byte addr
    const unsigned lbase1 = (unsigned)(uintptr_t)&Bsm[1][0];
    // global byte offset of this thread's 16B for chunk k: goff0 + 2*k
    const unsigned goff0 =
        (unsigned)((((unsigned)(n0 + sc)) * (unsigned)KTOT + (unsigned)(sp * 8)) * 2u);

#define ASYNC_B_LOAD(ldsaddr, goff)                                         \
    asm volatile("global_load_async_to_lds_b128 %0, %1, %2"                 \
                 :: "v"(ldsaddr), "v"(goff), "s"(Wt) : "memory")

    // ---- A-fragment (16x32 bf16) per-lane K groups ----
    //   lanes 0-15 : K = {k+0..7,  k+16..23}   (row M = l15)
    //   lanes 16-31: K = {k+8..15, k+24..31}
    const int g0 = half * 8;
    const int g1 = 16 + half * 8;
    const float* xrow = X + (size_t)(m0 + l15) * KTOT;

    // B-fragment per-lane LDS element offset: col*32 + half*8 (K groups half*8..+7, 16+half*8..+7)
    const int boff = (half * 8);

    v8f acc[4] = {};

    // prologue: stage chunk 0 into buffer 0
    ASYNC_B_LOAD(lbase0 + lds_in, goff0);

    const int NCHUNK = KTOT / 32;
    int buf = 0;
    for (int i = 0; i < NCHUNK; i++) {
        const int k = i * 32;

        // stage next chunk into the other buffer (safe: barrier at end of i-1
        // guaranteed all waves finished reading it)
        if (i + 1 < NCHUNK) {
            const unsigned nb = (buf ? lbase0 : lbase1);
            ASYNC_B_LOAD(nb + lds_in, goff0 + (unsigned)((k + 32) * 2));
            asm volatile("s_wait_asynccnt 0x1" ::: "memory");  // chunk i landed
        } else {
            asm volatile("s_wait_asynccnt 0x0" ::: "memory");
        }
        __syncthreads();   // all waves' async writes visible

        // ---- load A (16 fp32 per lane, two contiguous 32B chunks) ----
        float4 a0 = *(const float4*)(xrow + k + g0);
        float4 a1 = *(const float4*)(xrow + k + g0 + 4);
        float4 a2 = *(const float4*)(xrow + k + g1);
        float4 a3 = *(const float4*)(xrow + k + g1 + 4);
        float af[16] = {a0.x, a0.y, a0.z, a0.w, a1.x, a1.y, a1.z, a1.w,
                        a2.x, a2.y, a2.z, a2.w, a3.x, a3.y, a3.z, a3.w};

        // ---- split fp32 -> bf16 hi + bf16 lo ----
        union { unsigned short s[16]; v16bf v; } Ahi, Alo;
#pragma unroll
        for (int j = 0; j < 16; j++) {
            unsigned u  = __float_as_uint(af[j]);
            unsigned hb = u & 0xFFFF0000u;               // truncate to bf16
            float lf    = af[j] - __uint_as_float(hb);   // exact residual
            Ahi.s[j] = (unsigned short)(u >> 16);
            Alo.s[j] = (unsigned short)(__float_as_uint(lf) >> 16);
        }

        // ---- 4 N-subtiles from LDS, 2 chained WMMAs each (hi + lo) ----
        const unsigned short* bufp = &Bsm[buf][0];
#pragma unroll
        for (int t = 0; t < 4; t++) {
            const unsigned short* bp = bufp + (t * 16 + l15) * 32 + boff;
            union { uint4 q[2]; v16bf v; } Bf;
            Bf.q[0] = *(const uint4*)(bp);
            Bf.q[1] = *(const uint4*)(bp + 16);

            acc[t] = __builtin_amdgcn_wmma_f32_16x16x32_bf16(
                false, Ahi.v, false, Bf.v, (short)0, acc[t], false, false);
            acc[t] = __builtin_amdgcn_wmma_f32_16x16x32_bf16(
                false, Alo.v, false, Bf.v, (short)0, acc[t], false, false);
        }

        __syncthreads();   // protect buf before it is re-staged next iteration
        buf ^= 1;
    }
#undef ASYNC_B_LOAD

    // ---- epilogue: C/D layout VGPR r -> row m0 + r + half*8, col n0+t*16+l15 ----
#pragma unroll
    for (int t = 0; t < 4; t++) {
        const int col = n0 + t * 16 + l15;
        const float bv = bias[col];
#pragma unroll
        for (int r = 0; r < 8; r++) {
            const int row = m0 + r + half * 8;
            Out[(size_t)row * NTOT + col] = acc[t][r] + bv;
        }
    }
}

// --------------------------------- launcher ---------------------------------

extern "C" void kernel_launch(void* const* d_in, const int* in_sizes, int n_in,
                              void* d_out, int out_size, void* d_ws, size_t ws_size,
                              hipStream_t stream) {
    const float* X    = (const float*)d_in[0];   // [B,S,I] fp32
    const float* W    = (const float*)d_in[1];   // [O,I]   fp32
    const float* bias = (const float*)d_in[2];   // [O]     fp32
    float* Out        = (float*)d_out;

    const int nW = in_sizes[1];                  // O*I = 1M
    const int M  = in_sizes[0] / KTOT;           // B*S = 32768

    // workspace: [0,1KB) partials, [1KB,1KB+4) thresh, [4KB,...) bf16 weights
    float* partial        = (float*)d_ws;
    float* thresh         = partial + 256;
    unsigned short* Wt    = (unsigned short*)((char*)d_ws + 4096);

    abssum_partial<<<256, 256, 0, stream>>>(W, partial, nW);
    finalize_thresh<<<1, 256, 0, stream>>>(partial, thresh, 1.0f / (float)nW);
    quantize_w<<<(nW + 255) / 256, 256, 0, stream>>>(W, thresh, Wt, nW);

    dim3 grid(NTOT / 64, M / 128);
    gemm_ternary<<<grid, 256, 0, stream>>>(X, Wt, bias, Out);
}
